// HunyuanVideo15AttnBlock_28690381537994
// MI455X (gfx1250) — compile-verified, matrix operand order checked by reading, not run on hardware
//
#include <hip/hip_runtime.h>
#include <hip/hip_bf16.h>

// ---------------------------------------------------------------------------
// HunyuanVideo attn block on gfx1250 (wave32, WMMA f32_16x16x32_f16).
// Pipeline: [wconv] f32->f16 weights, [rmsnorm] xn[S,C] f16,
//           [qkv]  Q[S,C],K[S,C],Vt[C,S] f16 via WMMA GEMM (32x64 wave tiles),
//           [attn] flash attention, frame-causal mask folded into loop bound,
//                  K/V tiles staged with global_load_async_to_lds_b128,
//           [oproj] WMMA GEMM + bias + residual -> out[C,S] f32.
// Workspace layout (needs 34 MB):
//   [0,8M)   xn f16  (reused as attention output O after qkv)
//   [8,16M)  Q  f16
//   [16,24M) K  f16
//   [24,32M) Vt f16 (transposed [C,S])
//   [32,34M) wq,wk,wv,wo f16
// ---------------------------------------------------------------------------

typedef _Float16 h16;
typedef __attribute__((ext_vector_type(16))) _Float16 v16h;
typedef __attribute__((ext_vector_type(8)))  _Float16 v8h;
typedef __attribute__((ext_vector_type(8)))  float    v8f;
typedef __attribute__((ext_vector_type(4)))  float    v4f;

#define S_TOK 8192
#define CCH   512

// A/B operand chunking for V_WMMA_F32_16X16X32_F16 (wave32):
//   lane = (half?16:0) + m ; holds K elements {half*8 .. half*8+7} and
//   {half*8+16 .. half*8+23} relative to the 32-wide K step.
__device__ __forceinline__ v16h ld_ab(const h16* base) {
  v8h lo = *(const v8h*)(base);
  v8h hi = *(const v8h*)(base + 16);
  return __builtin_shufflevector(lo, hi, 0,1,2,3,4,5,6,7,8,9,10,11,12,13,14,15);
}

__device__ __forceinline__ v8f wmma16(v16h a, v16h b, v8f c) {
  return __builtin_amdgcn_wmma_f32_16x16x32_f16(false, a, false, b,
                                                (short)0, c, false, false);
}

// Async copy global -> LDS (ASYNCcnt tracked). lds_off = low 32 bits of the
// generic shared-aperture address (addr[31:0] == LDS byte offset per ISA).
__device__ __forceinline__ void async_ld_b128(unsigned lds_off, const void* gptr) {
  asm volatile("global_load_async_to_lds_b128 %0, %1, off"
               :: "v"(lds_off), "v"(gptr)
               : "memory");
}
__device__ __forceinline__ void wait_async0() {
  asm volatile("s_wait_asynccnt 0" ::: "memory");
}

// ---------------------------------------------------------------------------
// K0: convert the four 512x512 f32 weights to f16 (row-major [out_ch, in_ch]).
// ---------------------------------------------------------------------------
__global__ void __launch_bounds__(256) wconv_kernel(
    const float* __restrict__ wq, const float* __restrict__ wk,
    const float* __restrict__ wv, const float* __restrict__ wo,
    h16* __restrict__ wh) {
  int idx = blockIdx.x * blockDim.x + threadIdx.x;   // 4*C*C threads exactly
  int m   = idx >> 18;                               // / (512*512)
  int r   = idx & (CCH * CCH - 1);
  const float* src = (m == 0) ? wq : (m == 1) ? wk : (m == 2) ? wv : wo;
  wh[idx] = (h16)src[r];
}

// ---------------------------------------------------------------------------
// K1: RMSNorm over channels. x is [C,S]; one thread per token, coalesced
// channel-major reads; writes token-major xn[S,C] f16.
// ---------------------------------------------------------------------------
__global__ void __launch_bounds__(256) rmsnorm_kernel(
    const float* __restrict__ x, const float* __restrict__ gamma,
    h16* __restrict__ xn) {
  int s = blockIdx.x * blockDim.x + threadIdx.x;     // S_TOK threads exactly
  float ss = 0.f;
  for (int c = 0; c < CCH; ++c) {
    float v = x[(size_t)c * S_TOK + s];
    ss += v * v;
  }
  float nrm = sqrtf(ss);
  float sc  = 22.627416997969522f / fmaxf(nrm, 1e-12f);  // sqrt(C)/||x||
  for (int c = 0; c < CCH; ++c) {
    xn[(size_t)s * CCH + c] = (h16)(x[(size_t)c * S_TOK + s] * sc * gamma[c]);
  }
}

// ---------------------------------------------------------------------------
// K2: QKV GEMMs. Each wave computes a 32x64 tile of one projection
// (2 A row-tiles share 4 B tiles -> 8 WMMAs / 12 x 16B loads per K-step).
// A = xn[S,C], B column n = weight row (out_ch) n -> contiguous per lane.
// mat 0 -> Q[S,C], 1 -> K[S,C], 2 -> V stored transposed Vt[C,S]
// (D rows are contiguous in Vt -> one 16B store per lane per n-tile).
// ---------------------------------------------------------------------------
__global__ void __launch_bounds__(256) qkv_kernel(
    const h16* __restrict__ xn, const h16* __restrict__ wh,
    const float* __restrict__ bq, const float* __restrict__ bk,
    const float* __restrict__ bv,
    h16* __restrict__ Qh, h16* __restrict__ Kh, h16* __restrict__ Vt) {
  const int wid  = blockIdx.x * (blockDim.x >> 5) + (threadIdx.x >> 5);
  const int lane = threadIdx.x & 31;
  const int l15  = lane & 15;
  const int half = lane >> 4;
  const int mat  = wid >> 11;               // / (256*8)
  const int rem  = wid & 2047;
  const int m0   = (rem >> 3) << 5;         // 32-row tile
  const int n0   = (rem & 7) << 6;          // 64-wide out-channel group
  const h16* W = wh + (size_t)mat * CCH * CCH;
  const float* bias = (mat == 0) ? bq : (mat == 1) ? bk : bv;

  v8f zero = {};
  v8f acc[2][4];
#pragma unroll
  for (int t = 0; t < 2; ++t)
#pragma unroll
    for (int nt = 0; nt < 4; ++nt) acc[t][nt] = zero;

  const h16* aRow0 = xn + (size_t)(m0 + l15) * CCH + half * 8;
  const h16* aRow1 = xn + (size_t)(m0 + 16 + l15) * CCH + half * 8;
#pragma unroll
  for (int kc = 0; kc < 16; ++kc) {
    v16h a0 = ld_ab(aRow0 + kc * 32);
    v16h a1 = ld_ab(aRow1 + kc * 32);
#pragma unroll
    for (int nt = 0; nt < 4; ++nt) {
      v16h b = ld_ab(W + (size_t)(n0 + nt * 16 + l15) * CCH + kc * 32 + half * 8);
      acc[0][nt] = wmma16(a0, b, acc[0][nt]);
      acc[1][nt] = wmma16(a1, b, acc[1][nt]);
    }
  }

#pragma unroll
  for (int t = 0; t < 2; ++t) {
    const int rowbase = m0 + t * 16 + half * 8;
#pragma unroll
    for (int nt = 0; nt < 4; ++nt) {
      int col = n0 + nt * 16 + l15;
      float b = bias[col];
      if (mat < 2) {
        h16* dst = (mat == 0) ? Qh : Kh;
#pragma unroll
        for (int r = 0; r < 8; ++r)
          dst[(size_t)(rowbase + r) * CCH + col] = (h16)(acc[t][nt][r] + b);
      } else {
        v8h o;
#pragma unroll
        for (int r = 0; r < 8; ++r) o[r] = (h16)(acc[t][nt][r] + b);
        *(v8h*)(Vt + (size_t)col * S_TOK + rowbase) = o;   // contiguous 16B
      }
    }
  }
}

// ---------------------------------------------------------------------------
// K3: flash attention. Block = 4 waves = 64 query rows (64 | 1024 so the
// whole block is in one frame -> mask is just the key-loop bound).
// Per wave: Q tile pre-built as 16 A-operands (128 VGPRs), 16x512 f32
// accumulator (256 VGPRs). K tile then V tile staged through one shared
// 32KB LDS buffer via global_load_async_to_lds_b128 (ASYNCcnt);
// exp(P) bounced through 1KB LDS to re-layout C->A.
// ---------------------------------------------------------------------------
__global__ void __launch_bounds__(128) attn_kernel(
    const h16* __restrict__ Qh, const h16* __restrict__ Kh,
    const h16* __restrict__ Vt, h16* __restrict__ Oh) {
  __shared__ __align__(16) h16 SH[32 * CCH];      // 32KB: K tile, then V tile
  __shared__ __align__(16) h16 P[4][16 * 32];     // per-wave exp(S) bounce

  const int wid   = threadIdx.x >> 5;
  const int lane  = threadIdx.x & 31;
  const int l15   = lane & 15;
  const int half  = lane >> 4;
  const int qrow0 = blockIdx.x * 64 + wid * 16;
  const unsigned shbase = (unsigned)(uintptr_t)&SH[0];   // LDS byte offset

  // Preload Q tile as WMMA A operands for all 16 K-steps.
  v16h qa[16];
  const h16* qptr = Qh + (size_t)(qrow0 + l15) * CCH + half * 8;
#pragma unroll
  for (int kc = 0; kc < 16; ++kc) qa[kc] = ld_ab(qptr + kc * 32);

  v8f zero = {};
  v8f acc[32];
#pragma unroll
  for (int nt = 0; nt < 32; ++nt) acc[nt] = zero;
  float mrow[8], lrow[8];
#pragma unroll
  for (int r = 0; r < 8; ++r) { mrow[r] = -3.0e38f; lrow[r] = 0.f; }

  const float sm  = 0.044194173824159216f;        // 1/sqrt(512)
  const int frame = qrow0 >> 10;
  const int nkt   = (frame + 1) << 5;             // 32-key tiles allowed

  for (int kt = 0; kt < nkt; ++kt) {
    const int kb = kt * 32;
    __syncthreads();                              // prior tile reads done
    {                                             // async-stage K tile (32KB)
      const char* gK = (const char*)(Kh + (size_t)kb * CCH);
#pragma unroll
      for (int i = 0; i < 16; ++i) {
        int j = threadIdx.x + i * 128;
        async_ld_b128(shbase + j * 16, gK + (size_t)j * 16);
      }
    }
    if (kt + 1 < nkt)                             // global_prefetch_b8 next K tile
      __builtin_prefetch(Kh + (size_t)(kb + 32) * CCH + threadIdx.x * 128, 0, 0);
    wait_async0();
    __syncthreads();

    // S = Q * K^T : 16 queries x 32 keys (two 16x16 C tiles)
    v8f s0 = zero, s1 = zero;
#pragma unroll
    for (int kc = 0; kc < 16; ++kc) {
      v16h b0 = ld_ab(&SH[(size_t)l15 * CCH + kc * 32 + half * 8]);
      v16h b1 = ld_ab(&SH[(size_t)(16 + l15) * CCH + kc * 32 + half * 8]);
      s0 = wmma16(qa[kc], b0, s0);
      s1 = wmma16(qa[kc], b1, s1);
    }

    // Online softmax. C/D layout: VGPR r <-> row (r + half*8), lanes = cols.
    float alpha[8];
#pragma unroll
    for (int r = 0; r < 8; ++r) {
      float t0 = s0[r] * sm, t1 = s1[r] * sm;
      float rm = fmaxf(t0, t1);
      rm = fmaxf(rm, __shfl_xor(rm, 1));
      rm = fmaxf(rm, __shfl_xor(rm, 2));
      rm = fmaxf(rm, __shfl_xor(rm, 4));
      rm = fmaxf(rm, __shfl_xor(rm, 8));          // 16-lane row max
      float mn = fmaxf(mrow[r], rm);
      alpha[r] = __expf(mrow[r] - mn);
      float p0 = __expf(t0 - mn), p1 = __expf(t1 - mn);
      float rs = p0 + p1;
      rs += __shfl_xor(rs, 1); rs += __shfl_xor(rs, 2);
      rs += __shfl_xor(rs, 4); rs += __shfl_xor(rs, 8);
      lrow[r] = lrow[r] * alpha[r] + rs;
      mrow[r] = mn;
      int row = r + half * 8;
      P[wid][row * 32 + l15]      = (h16)p0;      // bounce to A layout via LDS
      P[wid][row * 32 + l15 + 16] = (h16)p1;
    }
#pragma unroll
    for (int nt = 0; nt < 32; ++nt)
#pragma unroll
      for (int r = 0; r < 8; ++r) acc[nt][r] *= alpha[r];

    __syncthreads();                              // all waves done with K tile
    {                                             // async-stage V: SH[cout*32+key]
#pragma unroll
      for (int i = 0; i < 16; ++i) {
        int j = threadIdx.x + i * 128;            // 0..2047 16B chunks
        int cout = j >> 2, part = j & 3;
        async_ld_b128(shbase + j * 16,
                      (const void*)(Vt + (size_t)cout * S_TOK + kb + part * 8));
      }
    }
    wait_async0();
    __syncthreads();

    // O += P * V : A = exp(S) 16x32, B = V tile columns = out channels.
    v16h pa = ld_ab(&P[wid][l15 * 32 + half * 8]);
#pragma unroll
    for (int nt = 0; nt < 32; ++nt) {
      v16h vb = ld_ab(&SH[(size_t)(nt * 16 + l15) * 32 + half * 8]);
      acc[nt] = wmma16(pa, vb, acc[nt]);
    }
  }

  // Normalize and store O[S,C] f16.
  float rinv[8];
#pragma unroll
  for (int r = 0; r < 8; ++r) rinv[r] = 1.0f / lrow[r];
#pragma unroll
  for (int nt = 0; nt < 32; ++nt)
#pragma unroll
    for (int r = 0; r < 8; ++r)
      Oh[(size_t)(qrow0 + half * 8 + r) * CCH + nt * 16 + l15] =
          (h16)(acc[nt][r] * rinv[r]);
}

// ---------------------------------------------------------------------------
// K4: out = wo * O + bo + x, written channel-major [C,S] (= [B,C,T,H,W]).
// 32x64 wave tiles; D rows are contiguous in the output -> vectorized stores.
// ---------------------------------------------------------------------------
__global__ void __launch_bounds__(256) oproj_kernel(
    const h16* __restrict__ Oh, const h16* __restrict__ Wo,
    const float* __restrict__ bo, const float* __restrict__ x,
    float* __restrict__ out) {
  const int wid  = blockIdx.x * (blockDim.x >> 5) + (threadIdx.x >> 5);
  const int lane = threadIdx.x & 31;
  const int l15  = lane & 15;
  const int half = lane >> 4;
  const int m0   = (wid >> 3) << 5;
  const int n0   = (wid & 7) << 6;

  v8f zero = {};
  v8f acc[2][4];
#pragma unroll
  for (int t = 0; t < 2; ++t)
#pragma unroll
    for (int nt = 0; nt < 4; ++nt) acc[t][nt] = zero;

  const h16* aRow0 = Oh + (size_t)(m0 + l15) * CCH + half * 8;
  const h16* aRow1 = Oh + (size_t)(m0 + 16 + l15) * CCH + half * 8;
#pragma unroll
  for (int kc = 0; kc < 16; ++kc) {
    v16h a0 = ld_ab(aRow0 + kc * 32);
    v16h a1 = ld_ab(aRow1 + kc * 32);
#pragma unroll
    for (int nt = 0; nt < 4; ++nt) {
      v16h b = ld_ab(Wo + (size_t)(n0 + nt * 16 + l15) * CCH + kc * 32 + half * 8);
      acc[0][nt] = wmma16(a0, b, acc[0][nt]);
      acc[1][nt] = wmma16(a1, b, acc[1][nt]);
    }
  }

#pragma unroll
  for (int t = 0; t < 2; ++t) {
#pragma unroll
    for (int nt = 0; nt < 4; ++nt) {
      int col = n0 + nt * 16 + l15;
      float b = bo[col];
      size_t base = (size_t)col * S_TOK + m0 + t * 16 + half * 8;
      v4f x0 = *(const v4f*)(x + base);
      v4f x1 = *(const v4f*)(x + base + 4);
      v4f o0, o1;
#pragma unroll
      for (int i = 0; i < 4; ++i) {
        o0[i] = acc[t][nt][i] + b + x0[i];
        o1[i] = acc[t][nt][i + 4] + b + x1[i];
      }
      *(v4f*)(out + base)     = o0;
      *(v4f*)(out + base + 4) = o1;
    }
  }
}

// ---------------------------------------------------------------------------
extern "C" void kernel_launch(void* const* d_in, const int* in_sizes, int n_in,
                              void* d_out, int out_size, void* d_ws, size_t ws_size,
                              hipStream_t stream) {
  (void)in_sizes; (void)n_in; (void)out_size; (void)ws_size;
  const float* x     = (const float*)d_in[0];
  const float* gamma = (const float*)d_in[1];
  const float* wq    = (const float*)d_in[2];
  const float* bq    = (const float*)d_in[3];
  const float* wk    = (const float*)d_in[4];
  const float* bk    = (const float*)d_in[5];
  const float* wv    = (const float*)d_in[6];
  const float* bv    = (const float*)d_in[7];
  const float* wo    = (const float*)d_in[8];
  const float* bo    = (const float*)d_in[9];
  float* out = (float*)d_out;

  char* ws = (char*)d_ws;
  h16* xn = (h16*)(ws);                          // 8MB, reused as O
  h16* Qh = (h16*)(ws + ((size_t)8  << 20));     // 8MB
  h16* Kh = (h16*)(ws + ((size_t)16 << 20));     // 8MB
  h16* Vt = (h16*)(ws + ((size_t)24 << 20));     // 8MB
  h16* Wh = (h16*)(ws + ((size_t)32 << 20));     // 2MB (4 x 512x512 f16)

  wconv_kernel  <<<4096, 256, 0, stream>>>(wq, wk, wv, wo, Wh);
  rmsnorm_kernel<<<S_TOK / 256, 256, 0, stream>>>(x, gamma, xn);
  // 3 mats x 256 row-tiles x 8 col-groups = 6144 waves / 8 per block
  qkv_kernel    <<<768, 256, 0, stream>>>(xn, Wh, bq, bk, bv, Qh, Kh, Vt);
  attn_kernel   <<<S_TOK / 64, 128, 0, stream>>>(Qh, Kh, Vt, /*O=*/xn);
  // 256 row-tiles x 8 col-groups = 2048 waves / 8 per block
  oproj_kernel  <<<256, 256, 0, stream>>>(xn, Wh + (size_t)3 * CCH * CCH, bo, x, out);
}
